// LatentThoughtTransformer_22247930593586
// MI455X (gfx1250) — matrix operand order repaired and checked by reading, not compiled, not run on hardware
//
#include <hip/hip_runtime.h>
#include <hip/hip_bf16.h>

// ---------------------------------------------------------------------------
// Latent Thought Transformer forward for MI455X (gfx1250).
// Persistent per-batch-tile workgroup: 16 batch elements per block, whole
// 32-step scan with activations resident in LDS; all GEMMs use
// v_wmma_f32_16x16x32_bf16 on bf16 L2-resident weights.
//  - B fragments: explicit addrspace(1) casts -> global_load_b128 (not flat)
//  - A fragments: bf16 mirrors of all GEMM inputs kept in LDS, so the inner
//    loop is two ds_load_b128 per fragment with zero conversion VALU.
//  - K compile-time: 4 k-tiles of A/B preloaded into distinct registers so
//    global loads pipeline ahead of the WMMA chain (partial loadcnt waits).
//
// Input flattening assumption (setup_inputs() dict order, recursive):
//  0 start_tokens(i32 B), 1 op_sequences(i32 B*32), 2 lengths(i32 B),
//  3 embed(256*128), 4 pe(512*128),
//  5..16  layer0: Wqkv,bqkv,Wo,bo,g1,be1,W1,bf1,W2,bf2,g2,be2
// 17..28  layer1: same
// 29 proj_W, 30 proj_b, 31 inj_W, 32 inj_b, 33 rm, 34 rv,
// 35 gate_W1(128x129), 36 gate_b1, 37 gate_W2(1x128), 38 gate_b2,
// 39 head_W(200x128), 40 head_b(200)
// ---------------------------------------------------------------------------

#define NTHREADS 256
#define NWAVES   8
#define BT       16          // batch tile per workgroup
#define D        128
#define DFF      256
#define TF       128
#define NV       200
#define NVPAD    208
#define MAXLEN   32

#define GAS __attribute__((address_space(1)))

typedef __attribute__((ext_vector_type(16))) __bf16        v16bf;
typedef __attribute__((ext_vector_type(8)))  float         v8f;
typedef __attribute__((ext_vector_type(4)))  unsigned int  u32x4;

union FragU {
    unsigned short s[16];
    u32x4          u[2];
    v16bf          v;
};

__device__ __forceinline__ unsigned short f2bf(float f) {
    unsigned u = __float_as_uint(f);
    unsigned r = (u + 0x7FFFu + ((u >> 16) & 1u)) >> 16;   // RNE
    return (unsigned short)r;
}
__device__ __forceinline__ float bf2f(unsigned short h) {
    return __uint_as_float(((unsigned)h) << 16);
}

struct Params {
    const int* start_tokens;
    const int* op_seq;
    const int* lengths;
    const float* embed;
    const float* pe;
    const float *bqkv[2], *bo[2], *g1[2], *be1[2], *bf1[2], *bf2[2], *g2[2], *be2[2];
    const float *proj_b, *inj_b, *rm, *rv, *gw1_full, *gate_b1, *gw2, *gate_b2, *head_b;
    // bf16 (u16 bits) weights in workspace, row-major (N, K)
    const unsigned short *wqkv[2], *wo[2], *w1[2], *w2[2];
    const unsigned short *wproj, *winj, *wgw1, *whead;
    float* out;
};

// --- WMMA fragment builders (ISA 7.12.2 layouts, wave32) --------------------
// A: 16x32 bf16 from LDS mirror. lanes 0-15: M=lane, K base 0; lanes 16-31:
// M=lane-16, K base 8.  s[0..7]=row[0..7], s[8..15]=row[16..23]  (contiguous
// 16B runs -> two ds_load_b128).
__device__ __forceinline__ v16bf load_a(const unsigned short* A, int ldA, int m0, int k0, int lane) {
    int m  = m0 + (lane & 15);
    int kb = (lane < 16) ? 0 : 8;
    const unsigned short* row = A + m * ldA + k0 + kb;
    FragU f;
    f.u[0] = *(const u32x4*)(row);
    f.u[1] = *(const u32x4*)(row + 16);
    return f.v;
}
// B: 32x16 bf16 from row-major global W(N,K). lanes 0-15: N=lane, K=0..7,16..23;
// lanes 16-31: N=lane-16, K=8..15,24..31.  Two global_load_b128 (addrspace(1)).
__device__ __forceinline__ v16bf load_b(const unsigned short* W, int K, int n0, int k0, int lane) {
    int n  = n0 + (lane & 15);
    int kb = (lane < 16) ? 0 : 8;
    const u32x4 GAS* p = (const u32x4 GAS*)(W + (size_t)n * K + k0 + kb);
    FragU f;
    f.u[0] = p[0];
    f.u[1] = p[2];
    return f.v;
}
// C/D: 16x16 f32, 8 VGPRs. lanes 0-15: M=r, N=lane; lanes 16-31: M=r+8, N=lane-16.
template <bool OUT_F32, bool OUT_B16, bool RELU>
__device__ __forceinline__ void store_c(v8f c, int m0, int n0, int lane, int nvalid,
                                        const float* bias,
                                        float* Of, int ldOf,
                                        unsigned short* Ob, int ldOb) {
    int n  = n0 + (lane & 15);
    int mb = m0 + ((lane < 16) ? 0 : 8);
    if (n < nvalid) {
        const float GAS* bg = (const float GAS*)bias;
        float bv = bias ? bg[n] : 0.0f;
#pragma unroll
        for (int r = 0; r < 8; ++r) {
            float v = c[r] + bv;
            if (RELU) v = fmaxf(v, 0.0f);
            if (OUT_F32) Of[(mb + r) * ldOf + n] = v;
            if (OUT_B16) Ob[(mb + r) * ldOb + n] = f2bf(v);
        }
    }
}

// GEMM: act(A(Mrows x K) @ W(N,K)^T + bias) -> f32 and/or bf16 LDS outputs.
template <int K, bool OUT_F32, bool OUT_B16, bool RELU>
__device__ void gemm(const unsigned short* A, int ldA, int Mrows,
                     const unsigned short* W, int Ncols, int nvalid,
                     const float* bias,
                     float* Of, int ldOf, unsigned short* Ob, int ldOb) {
    int lane = threadIdx.x & 31;
    int wave = threadIdx.x >> 5;
    int mt = Mrows >> 4, ntl = Ncols >> 4;
    int njobs = mt * ntl;
    for (int j = wave; j < njobs; j += NWAVES) {
        int m0 = (j / ntl) << 4;
        int n0 = (j % ntl) << 4;
        v8f c = {0.f, 0.f, 0.f, 0.f, 0.f, 0.f, 0.f, 0.f};
#pragma unroll
        for (int kc = 0; kc < K; kc += 128) {
            v16bf bf[4], af[4];
#pragma unroll
            for (int t = 0; t < 4; ++t)
                if (kc + 32 * t < K)
                    bf[t] = load_b(W, K, n0, kc + 32 * t, lane);
#pragma unroll
            for (int t = 0; t < 4; ++t)
                if (kc + 32 * t < K)
                    af[t] = load_a(A, ldA, m0, kc + 32 * t, lane);
#pragma unroll
            for (int t = 0; t < 4; ++t)
                if (kc + 32 * t < K)
                    c = __builtin_amdgcn_wmma_f32_16x16x32_bf16(false, af[t], false, bf[t],
                                                                (short)0, c, false, false);
        }
        store_c<OUT_F32, OUT_B16, RELU>(c, m0, n0, lane, nvalid, bias, Of, ldOf, Ob, ldOb);
    }
}

// --- attention (S<=3, H=4, DH=32), fp32 VALU -------------------------------
__device__ void attention(int S, const unsigned short* qkv, unsigned short* obuf, int tid) {
    const float scale = 0.17677669529663687f;   // 1/sqrt(32)
    int items = BT * 4 * S;
    for (int it = tid; it < items; it += NTHREADS) {
        int b = it / (4 * S);
        int rem = it % (4 * S);
        int h = rem / S, s = rem % S;
        const unsigned short* qr = qkv + (b * S + s) * 384 + h * 32;
        float sc[3];
        float mx = -1e30f;
        for (int t = 0; t < S; ++t) {
            const unsigned short* kr = qkv + (b * S + t) * 384 + 128 + h * 32;
            float d = 0.f;
#pragma unroll
            for (int e = 0; e < 32; ++e) d += bf2f(qr[e]) * bf2f(kr[e]);
            sc[t] = d * scale;
            mx = fmaxf(mx, sc[t]);
        }
        float sum = 0.f;
        for (int t = 0; t < S; ++t) { sc[t] = __expf(sc[t] - mx); sum += sc[t]; }
        float inv = 1.0f / sum;
#pragma unroll
        for (int e = 0; e < 32; ++e) {
            float acc = 0.f;
            for (int t = 0; t < S; ++t)
                acc += sc[t] * bf2f(qkv[(b * S + t) * 384 + 256 + h * 32 + e]);
            obuf[(b * S + s) * 128 + h * 32 + e] = f2bf(acc * inv);
        }
    }
}

// x[r] = LayerNorm(x[r] + tmp[r]) * g + be; refresh bf16 mirror xb.
// 4 lanes cooperate per row (wave32-safe shuffle groups).
__device__ void ln_residual(float* x, unsigned short* xb, const float* tmp, int rows,
                            const float* g, const float* be, int tid) {
    int grp = tid >> 2;
    int sub = tid & 3;
    for (int r = grp; r < rows; r += NTHREADS / 4) {
        float s = 0.f, s2 = 0.f;
        for (int d = sub; d < D; d += 4) {
            float v = x[r * D + d] + tmp[r * D + d];
            s += v; s2 += v * v;
        }
        s  += __shfl_xor(s, 1);  s  += __shfl_xor(s, 2);
        s2 += __shfl_xor(s2, 1); s2 += __shfl_xor(s2, 2);
        float m   = s * (1.0f / D);
        float var = s2 * (1.0f / D) - m * m;
        float inv = rsqrtf(var + 1e-5f);
        for (int d = sub; d < D; d += 4) {
            float v = x[r * D + d] + tmp[r * D + d];
            float y = (v - m) * inv * g[d] + be[d];
            x[r * D + d]  = y;
            xb[r * D + d] = f2bf(y);
        }
    }
}

// 2-layer post-norm encoder over 16*S rows resident in LDS
__device__ void run_encoder(const Params& P, int S, float* x, unsigned short* xb,
                            float* tmp, unsigned short* scr, unsigned short* obuf, int tid) {
    const int rows = BT * S;
    for (int l = 0; l < 2; ++l) {
        __syncthreads();
        // qkv = x @ Wqkv^T + bqkv  -> bf16 scratch (rows x 384)
        gemm<D, false, true, false>(xb, D, rows, P.wqkv[l], 384, 384, P.bqkv[l],
                                    nullptr, 0, scr, 384);
        __syncthreads();
        attention(S, scr, obuf, tid);
        __syncthreads();
        // o @ Wo^T + bo -> tmp (f32)
        gemm<D, true, false, false>(obuf, D, rows, P.wo[l], D, D, P.bo[l],
                                    tmp, D, nullptr, 0);
        __syncthreads();
        ln_residual(x, xb, tmp, rows, P.g1[l], P.be1[l], tid);
        __syncthreads();
        // f1 = relu(x @ W1^T + bf1) -> bf16 scratch (rows x 256)
        gemm<D, false, true, true>(xb, D, rows, P.w1[l], DFF, DFF, P.bf1[l],
                                   nullptr, 0, scr, DFF);
        __syncthreads();
        // f2 = f1 @ W2^T + bf2 -> tmp (f32)
        gemm<DFF, true, false, false>(scr, DFF, rows, P.w2[l], D, D, P.bf2[l],
                                      tmp, D, nullptr, 0);
        __syncthreads();
        ln_residual(x, xb, tmp, rows, P.g2[l], P.be2[l], tid);
    }
    __syncthreads();
}

// LDS partition (bytes): x,tmp,tf,nf,hid f32; scr,obuf,xb,tfb,nfb,poolb bf16; unc,gg
#define SMEM_BYTES ((48*128 + 48*128 + 3*16*128 + 2*16)*4 + \
                    (48*384 + 48*128 + 48*128 + 3*16*128)*2)

__global__ __launch_bounds__(NTHREADS, 2)
void ltt_kernel(Params P) {
    extern __shared__ char smem_raw[];
    float* x    = (float*)smem_raw;            // 48 x 128
    float* tmp  = x + 48 * 128;                // 48 x 128 (also 16x208 head out)
    float* tf   = tmp + 48 * 128;              // 16 x 128 thoughts (f32 master)
    float* nf   = tf + BT * TF;                // 16 x 128
    float* hid  = nf + BT * TF;                // 16 x 128
    float* unc  = hid + BT * D;                // 16
    float* gg   = unc + BT;                    // 16
    unsigned short* scr   = (unsigned short*)(gg + BT);  // 48 x 384 (qkv / f1)
    unsigned short* obuf  = scr + 48 * 384;    // 48 x 128
    unsigned short* xb    = obuf + 48 * 128;   // 48 x 128  bf16 mirror of x
    unsigned short* tfb   = xb + 48 * 128;     // 16 x 128  bf16 mirror of tf
    unsigned short* nfb   = tfb + BT * TF;     // 16 x 128  bf16 mirror of nf
    unsigned short* poolb = nfb + BT * TF;     // 16 x 128  pooled (bf16 only)

    const int tid  = threadIdx.x;
    const int b0   = blockIdx.x * BT;
    const int grp8 = tid >> 3, sub8 = tid & 7;

    const float GAS* embed = (const float GAS*)P.embed;
    const float GAS* pe    = (const float GAS*)P.pe;
    float GAS* outp        = (float GAS*)P.out;

    // warm L2 with layer-0 weights (global_prefetch_b8)
    for (int o = tid; o < (384 * 128 * 2) / 64; o += NTHREADS)
        __builtin_prefetch((const char*)P.wqkv[0] + o * 64, 0, 1);

    // ---- stage A: start token, S=1 ----
    for (int i = tid; i < BT * D; i += NTHREADS) {
        int b = i >> 7, d = i & 127;
        int tok = P.start_tokens[b0 + b];
        float v = embed[tok * D + d] + pe[d];
        x[i]  = v;
        xb[i] = f2bf(v);
    }
    run_encoder(P, 1, x, xb, tmp, scr, obuf, tid);
    // thoughts = h @ proj_W^T + proj_b  (f32 master + bf16 mirror)
    gemm<D, true, true, false>(xb, D, BT, P.wproj, TF, TF, P.proj_b, tf, TF, tfb, TF);
    __syncthreads();

    // ---- stage B: 32 scan steps ----
    for (int step = 0; step < MAXLEN; ++step) {
        // te = thoughts(16,2,64) @ inj_W^T + inj_b -> tmp (32 x 128)
        gemm<64, true, false, false>(tfb, 64, 2 * BT, P.winj, D, D, P.inj_b,
                                     tmp, D, nullptr, 0);
        __syncthreads();
        // combined = [te0+pe0, te1+pe1, emb(op)+pe2]
        for (int i = tid; i < BT * 3 * D; i += NTHREADS) {
            int d = i & 127, rs = i >> 7;
            int b = rs / 3, s = rs % 3;
            float v;
            if (s < 2) v = tmp[(b * 2 + s) * D + d] + pe[s * D + d];
            else {
                int tok = P.op_seq[(b0 + b) * MAXLEN + step];
                v = embed[tok * D + d] + pe[2 * D + d];
            }
            x[rs * D + d]  = v;
            xb[rs * D + d] = f2bf(v);
        }
        run_encoder(P, 3, x, xb, tmp, scr, obuf, tid);
        for (int i = tid; i < BT * D; i += NTHREADS) {
            int b = i >> 7, d = i & 127;
            float v = (x[(b * 3) * D + d] + x[(b * 3 + 1) * D + d] +
                       x[(b * 3 + 2) * D + d]) * (1.0f / 3.0f);
            poolb[i] = f2bf(v);
        }
        __syncthreads();
        gemm<D, true, true, false>(poolb, D, BT, P.wproj, TF, TF, P.proj_b,
                                   nf, TF, nfb, TF);
        __syncthreads();
        // unc[b] = mean(((nf-rm)^2)/(rv+1e-6))  -- 8 lanes per row
        if (grp8 < BT) {
            float s = 0.f;
            for (int d = sub8; d < TF; d += 8) {
                float dv = nf[grp8 * TF + d] - P.rm[d];
                s += dv * dv / (P.rv[d] + 1e-6f);
            }
            s += __shfl_xor(s, 1); s += __shfl_xor(s, 2); s += __shfl_xor(s, 4);
            if (sub8 == 0) unc[grp8] = s * (1.0f / TF);
        }
        __syncthreads();
        // gate hidden: nf @ W1[:, :128]^T + b1 (+ unc * W1[:,128] afterwards)
        gemm<TF, true, false, false>(nfb, TF, BT, P.wgw1, D, D, P.gate_b1,
                                     hid, D, nullptr, 0);
        __syncthreads();
        {
            const float GAS* gw1 = (const float GAS*)P.gw1_full;
            for (int i = tid; i < BT * D; i += NTHREADS) {
                int b = i >> 7, n = i & 127;
                float v = hid[i] + gw1[n * 129 + 128] * unc[b];
                hid[i] = fmaxf(v, 0.0f);
            }
        }
        __syncthreads();
        // g[b] = sigmoid(hid[b] . gw2 + b2)  -- 8 lanes per row
        if (grp8 < BT) {
            const float GAS* gw2 = (const float GAS*)P.gw2;
            float s = 0.f;
            for (int d = sub8; d < D; d += 8) s += hid[grp8 * D + d] * gw2[d];
            s += __shfl_xor(s, 1); s += __shfl_xor(s, 2); s += __shfl_xor(s, 4);
            if (sub8 == 0) gg[grp8] = 1.0f / (1.0f + __expf(-(s + P.gate_b2[0])));
        }
        __syncthreads();
        for (int i = tid; i < BT * TF; i += NTHREADS) {
            int b = i >> 7;
            float g   = gg[b];
            float cur = tf[i];
            float up  = g * nf[i] + (1.0f - g) * cur;
            bool act  = step < P.lengths[b0 + b];
            float nv  = act ? up : cur;
            tf[i]  = nv;
            tfb[i] = f2bf(nv);
        }
        __syncthreads();
    }

    // ---- stage C: final readout, S=2 ----
    gemm<64, true, false, false>(tfb, 64, 2 * BT, P.winj, D, D, P.inj_b,
                                 tmp, D, nullptr, 0);
    __syncthreads();
    for (int i = tid; i < BT * 2 * D; i += NTHREADS) {
        int d = i & 127, rs = i >> 7, s = rs & 1;
        float v = tmp[i] + pe[s * D + d];
        x[i]  = v;
        xb[i] = f2bf(v);
    }
    run_encoder(P, 2, x, xb, tmp, scr, obuf, tid);
    for (int i = tid; i < BT * D; i += NTHREADS) {
        int b = i >> 7, d = i & 127;
        poolb[i] = f2bf(0.5f * (x[(b * 2) * D + d] + x[(b * 2 + 1) * D + d]));
    }
    __syncthreads();
    // logits = pooled @ head_W^T + head_b  (N padded to 208, 200 valid)
    gemm<D, true, false, false>(poolb, D, BT, P.whead, NVPAD, NV, P.head_b,
                                tmp, NVPAD, nullptr, 0);
    __syncthreads();
    for (int i = tid; i < BT * NV; i += NTHREADS) {
        int b = i / NV, n = i % NV;
        outp[(size_t)(b0 + b) * NV + n] = tmp[b * NVPAD + n];
    }
}

// fp32 -> bf16 weight conversion (with optional row padding + strided src)
__global__ void cvt_bf16(const float* __restrict__ src, unsigned short* __restrict__ dst,
                         int rows, int cols, int src_ld, int dst_rows) {
    int i = blockIdx.x * blockDim.x + threadIdx.x;
    int total = dst_rows * cols;
    if (i >= total) return;
    int r = i / cols, c = i % cols;
    const float GAS* s = (const float GAS*)src;
    unsigned short GAS* d = (unsigned short GAS*)dst;
    d[i] = (r < rows) ? f2bf(s[r * src_ld + c]) : (unsigned short)0;
}

extern "C" void kernel_launch(void* const* d_in, const int* in_sizes, int n_in,
                              void* d_out, int out_size, void* d_ws, size_t ws_size,
                              hipStream_t stream) {
    (void)in_sizes; (void)n_in; (void)out_size; (void)ws_size;
    const float* F[48];
    for (int i = 0; i < 41; ++i) F[i] = (const float*)d_in[i];

    unsigned short* ws = (unsigned short*)d_ws;
    size_t off = 0;
    auto alloc = [&](size_t n) { unsigned short* p = ws + off; off += n; return p; };

    unsigned short *WQKV[2], *WO[2], *W1B[2], *W2B[2];
    for (int l = 0; l < 2; ++l) {
        WQKV[l] = alloc(384 * 128);
        WO[l]   = alloc(128 * 128);
        W1B[l]  = alloc(256 * 128);
        W2B[l]  = alloc(128 * 256);
    }
    unsigned short* WPROJ = alloc(128 * 128);
    unsigned short* WINJ  = alloc(128 * 64);
    unsigned short* WGW1  = alloc(128 * 128);
    unsigned short* WHEAD = alloc(208 * 128);

    auto cvt = [&](const float* src, unsigned short* dst, int rows, int cols,
                   int src_ld, int dst_rows) {
        int total = dst_rows * cols;
        cvt_bf16<<<(total + 255) / 256, 256, 0, stream>>>(src, dst, rows, cols, src_ld, dst_rows);
    };
    for (int l = 0; l < 2; ++l) {
        int base = 5 + 12 * l;
        cvt(F[base + 0], WQKV[l], 384, 128, 128, 384);
        cvt(F[base + 2], WO[l],   128, 128, 128, 128);
        cvt(F[base + 6], W1B[l],  256, 128, 128, 256);
        cvt(F[base + 8], W2B[l],  128, 256, 256, 128);
    }
    cvt(F[29], WPROJ, 128, 128, 128, 128);
    cvt(F[31], WINJ,  128,  64,  64, 128);
    cvt(F[35], WGW1,  128, 128, 129, 128);   // first 128 cols of gate_W1
    cvt(F[39], WHEAD, 200, 128, 128, 208);   // pad head rows 200->208 with zeros

    Params P;
    P.start_tokens = (const int*)d_in[0];
    P.op_seq       = (const int*)d_in[1];
    P.lengths      = (const int*)d_in[2];
    P.embed = F[3];
    P.pe    = F[4];
    for (int l = 0; l < 2; ++l) {
        int base = 5 + 12 * l;
        P.wqkv[l] = WQKV[l]; P.bqkv[l] = F[base + 1];
        P.wo[l]   = WO[l];   P.bo[l]   = F[base + 3];
        P.g1[l]   = F[base + 4]; P.be1[l] = F[base + 5];
        P.w1[l]   = W1B[l];  P.bf1[l]  = F[base + 7];
        P.w2[l]   = W2B[l];  P.bf2[l]  = F[base + 9];
        P.g2[l]   = F[base + 10]; P.be2[l] = F[base + 11];
    }
    P.proj_b = F[30]; P.inj_b = F[32];
    P.rm = F[33]; P.rv = F[34];
    P.gw1_full = F[35]; P.gate_b1 = F[36];
    P.gw2 = F[37]; P.gate_b2 = F[38];
    P.head_b = F[40];
    P.wproj = WPROJ; P.winj = WINJ; P.wgw1 = WGW1; P.whead = WHEAD;
    P.out = (float*)d_out;

    ltt_kernel<<<2048 / BT, NTHREADS, SMEM_BYTES, stream>>>(P);
}
